// AttentionModule_76321568850471
// MI455X (gfx1250) — matrix operand order repaired and checked by reading, not compiled, not run on hardware
//
#include <hip/hip_runtime.h>

typedef __attribute__((ext_vector_type(16))) __bf16        v16bf;
typedef __attribute__((ext_vector_type(8)))  float         v8f;
typedef __attribute__((ext_vector_type(8)))  unsigned int  v8u;

#define C_IN   512
#define C_RED  64
#define NB     16
#define NPIX   4096   // 64*64

// ---------- bf16 helpers ----------
__device__ __forceinline__ unsigned short f2bf(float f) {   // RNE, for stored tensors
  unsigned int u = __builtin_bit_cast(unsigned int, f);
  u += 0x7FFFu + ((u >> 16) & 1u);
  return (unsigned short)(u >> 16);
}
__device__ __forceinline__ __bf16 bfbits(unsigned short s) {
  return __builtin_bit_cast(__bf16, s);
}
// one v_perm_b32 packs two fp32 -> two bf16 (truncate)
__device__ __forceinline__ unsigned int pack2_bf16(float lo, float hi) {
  return __builtin_amdgcn_perm(__builtin_bit_cast(unsigned int, hi),
                               __builtin_bit_cast(unsigned int, lo),
                               0x07060302u);
}

// ---------- WMMA wrapper ----------
__device__ __forceinline__ v8f wmma_bf16(v16bf a, v16bf b, v8f c) {
  return __builtin_amdgcn_wmma_f32_16x16x32_bf16(false, a, false, b,
                                                 (short)0, c, false, false);
}

// ---------- fragment loaders (ISA 7.12.2, wave32) ----------
// A 16x32 bf16: lane (h=lane/16) holds k in [16h,16h+16)
struct BfPair { uint4 lo, hi; };
__device__ __forceinline__ v16bf load_a_bf16(const unsigned short* row, int h) {
  BfPair p;
  p.lo = *(const uint4*)(row + h * 16);
  p.hi = *(const uint4*)(row + h * 16 + 8);
  return __builtin_bit_cast(v16bf, p);
}
// B 32x16 bf16: lane holds col n=lane%16, k = 16*(lane/16)+e  (one 32B load)
__device__ __forceinline__ v16bf load_b_bf16(const unsigned short* krun) {
  return *(const v16bf*)krun;
}
// B fragment from fp32 strided source, perm-packed
__device__ __forceinline__ v16bf load_b_f32_strided(const float* base, int stride) {
  v8u u;
#pragma unroll
  for (int e = 0; e < 8; ++e)
    u[e] = pack2_bf16(base[(2 * e) * stride], base[(2 * e + 1) * stride]);
  return __builtin_bit_cast(v16bf, u);
}

// ============ kernel 0: convert w1 (64x512) and w2 (512x64) to bf16 ======
__global__ void k_prep(const float* __restrict__ w1, const float* __restrict__ w2,
                       unsigned short* __restrict__ w1b, unsigned short* __restrict__ w2b) {
  int i = blockIdx.x * 256 + threadIdx.x;               // 65536 threads
  if (i < C_RED * C_IN) w1b[i] = f2bf(w1[i]);
  else                  w2b[i - C_RED * C_IN] = f2bf(w2[i - C_RED * C_IN]);
}

// ============ kernel 1: q = w1 @ x + b1  (bf16: q row-major AND qT) ======
// grid (32, 16): x=column-tile (128 cols/block), y=batch.  256 thr = 8 waves.
__global__ void k_qproj(const float* __restrict__ x, const unsigned short* __restrict__ w1b,
                        const float* __restrict__ b1, unsigned short* __restrict__ q,
                        unsigned short* __restrict__ qt) {
  const int tid  = threadIdx.x;
  const int wave = tid >> 5, lane = tid & 31;
  const int h = lane >> 4, ln = lane & 15;
  const int batch = blockIdx.y;
  const int col   = blockIdx.x * 128 + wave * 16 + ln;

  const float* xb = x + (size_t)batch * C_IN * NPIX;
  v8f acc[4];
#pragma unroll
  for (int mt = 0; mt < 4; ++mt) acc[mt] = (v8f){0.f,0.f,0.f,0.f,0.f,0.f,0.f,0.f};

  for (int k0 = 0; k0 < C_IN; k0 += 32) {
    v16bf bf = load_b_f32_strided(xb + (size_t)(k0 + h * 16) * NPIX + col, NPIX);
    if (k0 + 32 < C_IN)   // prefetch next k-slab of x (global_prefetch_b8)
      __builtin_prefetch(xb + (size_t)(k0 + 32 + h * 16) * NPIX + col, 0, 3);
#pragma unroll
    for (int mt = 0; mt < 4; ++mt) {
      v16bf af = load_a_bf16(w1b + (mt * 16 + ln) * C_IN + k0, h);
      acc[mt] = wmma_bf16(af, bf, acc[mt]);
    }
  }
  unsigned short* qb  = q  + (size_t)batch * C_RED * NPIX;
  unsigned short* qtb = qt + ((size_t)batch * NPIX + col) * C_RED;
#pragma unroll
  for (int mt = 0; mt < 4; ++mt) {
    unsigned short s[8];
#pragma unroll
    for (int r = 0; r < 8; ++r) {
      int m = mt * 16 + 8 * h + r;
      s[r] = f2bf(acc[mt][r] + b1[m]);
      qb[(size_t)m * NPIX + col] = s[r];            // q[c][n]
    }
    uint4 pk;                                        // qT[n][c] : 16B run
    pk.x = (unsigned int)s[0] | ((unsigned int)s[1] << 16);
    pk.y = (unsigned int)s[2] | ((unsigned int)s[3] << 16);
    pk.z = (unsigned int)s[4] | ((unsigned int)s[5] << 16);
    pk.w = (unsigned int)s[6] | ((unsigned int)s[7] << 16);
    *(uint4*)(qtb + mt * 16 + 8 * h) = pk;
  }
}

// ============ kernel 2: att = softmax(q @ q^T)  (bf16 out) ===============
// one block per batch; each of 8 waves OWNS two 16x16 tiles over full K=4096
// (tiles 2w and 2w+1 share the same M-tile -> one shared A fragment).
__global__ void k_att(const unsigned short* __restrict__ q,
                      unsigned short* __restrict__ att) {
  __shared__ float atts[C_RED * C_RED];
  const int tid = threadIdx.x;
  const int wave = tid >> 5, lane = tid & 31;
  const int h = lane >> 4, ln = lane & 15;
  const int batch = blockIdx.x;
  const unsigned short* qb = q + (size_t)batch * C_RED * NPIX;

  const int mt  = (2 * wave) >> 2;        // shared M-tile
  const int nt0 = (2 * wave) & 3, nt1 = nt0 + 1;

  v8f acc0 = (v8f){0.f,0.f,0.f,0.f,0.f,0.f,0.f,0.f};
  v8f acc1 = acc0;

  for (int k0 = 0; k0 < NPIX; k0 += 32) {
    v16bf af  = load_a_bf16(qb + (size_t)(mt  * 16 + ln) * NPIX + k0, h);
    v16bf bf0 = load_b_bf16(qb + (size_t)(nt0 * 16 + ln) * NPIX + k0 + h * 16);
    v16bf bf1 = load_b_bf16(qb + (size_t)(nt1 * 16 + ln) * NPIX + k0 + h * 16);
    acc0 = wmma_bf16(af, bf0, acc0);
    acc1 = wmma_bf16(af, bf1, acc1);
  }
#pragma unroll
  for (int r = 0; r < 8; ++r) {
    int m = mt * 16 + 8 * h + r;
    atts[m * C_RED + nt0 * 16 + ln] = acc0[r];
    atts[m * C_RED + nt1 * 16 + ln] = acc1[r];
  }
  __syncthreads();

  // row softmax (threads 0..63 each own one row), write bf16
  if (tid < C_RED) {
    float* row = atts + tid * C_RED;
    float mx = -3.0e38f;
    for (int j = 0; j < C_RED; ++j) mx = fmaxf(mx, row[j]);
    float s = 0.f;
    for (int j = 0; j < C_RED; ++j) { float e = __expf(row[j] - mx); row[j] = e; s += e; }
    float inv = 1.f / s;
    unsigned short* ab = att + (size_t)batch * C_RED * C_RED + tid * C_RED;
    for (int j = 0; j < C_RED; ++j) ab[j] = f2bf(row[j] * inv);
  }
}

// ====== kernel 3: o = att@qT ; y = w2@o + b2 + x  (fused) ================
// grid (128, 16): x = 32-column tile, y = batch. 256 thr = 8 waves.
__global__ void k_out(const unsigned short* __restrict__ qt,
                      const unsigned short* __restrict__ att,
                      const unsigned short* __restrict__ w2b, const float* __restrict__ b2,
                      const float* __restrict__ x, float* __restrict__ y) {
  __shared__ __align__(32) unsigned short o_lds[32 * C_RED];  // [col][row]
  const int tid = threadIdx.x;
  const int wave = tid >> 5, lane = tid & 31;
  const int h = lane >> 4, ln = lane & 15;
  const int batch = blockIdx.y;
  const int col0 = blockIdx.x * 32;

  const unsigned short* qtb = qt + (size_t)batch * NPIX * C_RED;
  const unsigned short* ab  = att + (size_t)batch * C_RED * C_RED;

  // phase 1: wave w -> row-strip s = w>>1, col-group g = w&1 (16 cols each).
  // B fragments are contiguous 32B runs of qT (k = channel dim).
  {
    const int s = wave >> 1, g = wave & 1;
    const int col = col0 + g * 16 + ln;
    v8f acc = (v8f){0.f,0.f,0.f,0.f,0.f,0.f,0.f,0.f};
#pragma unroll
    for (int kt = 0; kt < 2; ++kt) {
      const int k0 = kt * 32;
      v16bf af = load_a_bf16(ab + (s * 16 + ln) * C_RED + k0, h);
      v16bf bf = load_b_bf16(qtb + (size_t)col * C_RED + k0 + h * 16);
      acc = wmma_bf16(af, bf, acc);
    }
#pragma unroll
    for (int r = 0; r < 8; ++r) {
      int m = s * 16 + 8 * h + r;
      o_lds[(g * 16 + ln) * C_RED + m] = f2bf(acc[r]);
    }
  }
  __syncthreads();

  // phase 2: each wave produces 64 output rows (4 M-tiles) x 32 cols, K=64
  const int row_base = wave * 64;
  v8f acc2[4][2];
#pragma unroll
  for (int mt = 0; mt < 4; ++mt)
#pragma unroll
    for (int g = 0; g < 2; ++g)
      acc2[mt][g] = (v8f){0.f,0.f,0.f,0.f,0.f,0.f,0.f,0.f};
#pragma unroll
  for (int kt = 0; kt < 2; ++kt) {
    const int k0 = kt * 32;
    v16bf bf0 = load_b_bf16(&o_lds[ln * C_RED + k0 + h * 16]);
    v16bf bf1 = load_b_bf16(&o_lds[(16 + ln) * C_RED + k0 + h * 16]);
#pragma unroll
    for (int mt = 0; mt < 4; ++mt) {
      v16bf af = load_a_bf16(w2b + (row_base + mt * 16 + ln) * C_RED + k0, h);
      acc2[mt][0] = wmma_bf16(af, bf0, acc2[mt][0]);
      acc2[mt][1] = wmma_bf16(af, bf1, acc2[mt][1]);
    }
  }
#pragma unroll
  for (int mt = 0; mt < 4; ++mt)
#pragma unroll
    for (int g = 0; g < 2; ++g)
#pragma unroll
      for (int r = 0; r < 8; ++r) {
        int m = row_base + mt * 16 + 8 * h + r;
        size_t idx = ((size_t)batch * C_IN + m) * NPIX + col0 + g * 16 + ln;
        y[idx] = acc2[mt][g][r] + b2[m] + x[idx];
      }
}

// ============================ launcher ===================================
extern "C" void kernel_launch(void* const* d_in, const int* in_sizes, int n_in,
                              void* d_out, int out_size, void* d_ws, size_t ws_size,
                              hipStream_t stream) {
  const float* x  = (const float*)d_in[0];
  const float* w1 = (const float*)d_in[1];
  const float* b1 = (const float*)d_in[2];
  const float* w2 = (const float*)d_in[3];
  const float* b2 = (const float*)d_in[4];
  float* out = (float*)d_out;

  unsigned short* q_ws   = (unsigned short*)d_ws;                 // 16*64*4096 bf16 = 8 MB
  unsigned short* qt_ws  = q_ws   + (size_t)NB * C_RED * NPIX;    // 16*4096*64 bf16 = 8 MB
  unsigned short* att_ws = qt_ws  + (size_t)NB * NPIX * C_RED;    // 16*64*64 bf16
  unsigned short* w1b    = att_ws + (size_t)NB * C_RED * C_RED;   // 64*512 bf16
  unsigned short* w2b    = w1b    + (size_t)C_RED * C_IN;         // 512*64 bf16

  k_prep <<<dim3(256),     256, 0, stream>>>(w1, w2, w1b, w2b);
  k_qproj<<<dim3(32, NB),  256, 0, stream>>>(x, w1b, b1, q_ws, qt_ws);
  k_att  <<<dim3(NB),      256, 0, stream>>>(q_ws, att_ws);
  k_out  <<<dim3(128, NB), 256, 0, stream>>>(qt_ws, att_ws, w2b, b2, x, out);
}
